// SE_DiSCO_37520834298478
// MI455X (gfx1250) — matrix operand order repaired and checked by reading, not compiled, not run on hardware
//
#include <hip/hip_runtime.h>
#include <hip/hip_bf16.h>
#include <math.h>

// ---------------------------------------------------------------------------
// SE_DiSCO spectral-encoder pipeline for MI455X (gfx1250, wave32, WMMA).
// Matmul-shaped work (convs as implicit GEMM, SIREN/FiLM/to_z GEMMs) runs on
// v_wmma_f32_16x16x32_f16 with f16 operands staged through LDS via the CDNA5
// async global->LDS path. DFT / spectral pointwise work stays VALU f32
// (negligible FLOPs, precision-sensitive).
// ---------------------------------------------------------------------------

typedef _Float16 h16;
typedef __attribute__((ext_vector_type(8)))  _Float16 v8h;
typedef __attribute__((ext_vector_type(16))) _Float16 v16h;
typedef __attribute__((ext_vector_type(8)))  float    v8f;
typedef __attribute__((ext_vector_type(4)))  int      i32x4;

static constexpr int   BN      = 128;
static constexpr int   WIDTH   = 64;
static constexpr int   OUT_LEN = 301;
static constexpr int   SHD     = 256;
static constexpr int   NLAY    = 6;
static constexpr int   NBLK    = 4;
static constexpr float OMEGA0  = 30.0f;
static constexpr float PI32    = 3.14159265358979f;

#define DEVFN static __device__ __forceinline__

DEVFN float geluf(float x) { return 0.5f * x * (1.0f + erff(x * 0.70710678118f)); }

DEVFN v16h cat8(v8h lo, v8h hi) {
  return __builtin_shufflevector(lo, hi, 0, 1, 2, 3, 4, 5, 6, 7,
                                 8, 9, 10, 11, 12, 13, 14, 15);
}

// ---- async global->LDS copy (CDNA5), with compile-safe fallback -----------
#if __has_builtin(__builtin_amdgcn_global_load_async_to_lds_b128)
#define HAVE_ASYNC_CP 1
#endif

DEVFN void cp16B_to_lds(const h16* g, h16* l) {
#ifdef HAVE_ASYNC_CP
  __builtin_amdgcn_global_load_async_to_lds_b128(
      (__attribute__((address_space(1))) i32x4*)g,
      (__attribute__((address_space(3))) i32x4*)l, 0, 0);
#else
  *(v8h*)l = *(const v8h*)g;            // ds_store_b128 fallback
#endif
}

DEVFN void wait_cp() {
#ifdef HAVE_ASYNC_CP
#if __has_builtin(__builtin_amdgcn_s_wait_asynccnt)
  __builtin_amdgcn_s_wait_asynccnt(0);
#else
  asm volatile("s_wait_asynccnt 0" ::: "memory");
#endif
#endif
}

// ===========================================================================
// WMMA fragment conventions (CDNA5 ISA 7.12.2, wave32):
//   A (16x32 f16): lane l holds row M=l&15; K = kb+{0..7}, kb+16+{0..7},
//                  kb = (l<16) ? 0 : 8.
//   B (32x16 f16): lane l holds column N=l&15; element e = row K, same K order as A.
//   C/D (16x16 f32): lane l holds column N=l&15, VGPR r = row ((l>>4)*8+r).
// Block = 4 waves; each wave owns a 16(M)x32(N) tile (2 accumulators); the
// 32x32 f16 B panel per K-step is staged once per block into LDS (double
// buffered, async copy).
// ===========================================================================

// ------------------------- implicit-GEMM convolution -----------------------
__global__ __launch_bounds__(128) void conv_wmma(
    const h16* __restrict__ in, const h16* __restrict__ wp, float* __restrict__ out,
    int Cin, int H, int W, int Cout, int Ho, int Wo,
    int ks, int pad, int stride, int Kpad)
{
  __shared__ __align__(16) h16 panel[2][32][32];
  const int lane = threadIdx.x;
  const int tid  = threadIdx.y * 32 + lane;
  const int HoWo = Ho * Wo;
  int mBase = (blockIdx.x * 4 + threadIdx.y) * 16;
  if (mBase + 16 > HoWo) mBase = HoWo - 16;     // clamp; keeps barriers uniform
  const int nBase = blockIdx.y * 32;
  const int b = blockIdx.z;
  const int m  = mBase + (lane & 15);
  const int oy = m / Wo, ox = m % Wo;
  const int kb = (lane < 16) ? 0 : 8;
  const int ksq = ks * ks;
  const h16* inB = in + (size_t)b * Cin * H * W;
  const int srow = tid >> 2, scol = (tid & 3) * 8;   // staging: 16B per thread

  cp16B_to_lds(wp + (size_t)srow * Cout + nBase + scol, &panel[0][srow][scol]);

  v8f acc0 = {}, acc1 = {};
  int s = 0;
  for (int k0 = 0; k0 < Kpad; k0 += 32, s ^= 1) {
    wait_cp();
    __syncthreads();
    if (k0 + 32 < Kpad)
      cp16B_to_lds(wp + (size_t)(k0 + 32 + srow) * Cout + nBase + scol,
                   &panel[s ^ 1][srow][scol]);
    // A fragment: per-lane gather with implicit im2col + zero padding
    v16h a;
#pragma unroll
    for (int e = 0; e < 16; ++e) {
      int kk  = k0 + kb + ((e < 8) ? e : (8 + e));
      int ci  = kk / ksq;
      int rem = kk - ci * ksq;
      int ky  = rem / ks, kx = rem - ky * ks;
      int iy  = oy * stride - pad + ky;
      int ix  = ox * stride - pad + kx;
      h16 v = (h16)0.0f;
      if (ci < Cin && (unsigned)iy < (unsigned)H && (unsigned)ix < (unsigned)W)
        v = inB[((size_t)ci * H + iy) * W + ix];
      a[e] = v;
    }
    v16h b0, b1;
#pragma unroll
    for (int e = 0; e < 16; ++e) {
      int kr = kb + ((e < 8) ? e : (8 + e));
      b0[e] = panel[s][kr][lane & 15];
      b1[e] = panel[s][kr][16 + (lane & 15)];
    }
    acc0 = __builtin_amdgcn_wmma_f32_16x16x32_f16(false, a, false, b0,
                                                  (short)0, acc0, false, false);
    acc1 = __builtin_amdgcn_wmma_f32_16x16x32_f16(false, a, false, b1,
                                                  (short)0, acc1, false, false);
  }
  const int n0   = nBase + (lane & 15);
  const int mrow = mBase + ((lane >> 4) << 3);
  float* o0 = out + ((size_t)b * Cout + n0) * HoWo;
  float* o1 = o0 + (size_t)16 * HoWo;
#pragma unroll
  for (int r = 0; r < 8; ++r) { o0[mrow + r] = acc0[r]; o1[mrow + r] = acc1[r]; }
}

// ----------------------------- generic WMMA GEMM ---------------------------
enum { EP_STORE = 0, EP_BIAS = 1, EP_BIAS_GELU = 2, EP_FILM_SIN = 3 };

__global__ __launch_bounds__(128) void gemm_wmma(
    const h16* __restrict__ A, const h16* __restrict__ Bp,
    float* __restrict__ C, h16* __restrict__ Ch,
    int M, int N, int K,
    const float* __restrict__ bias,
    const float* __restrict__ film, int rows_per_b, int ep)
{
  __shared__ __align__(16) h16 panel[2][32][32];
  const int lane = threadIdx.x;
  const int tid  = threadIdx.y * 32 + lane;
  int mBase = (blockIdx.x * 4 + threadIdx.y) * 16;
  if (mBase + 16 > M) mBase = M - 16;           // clamp; keeps barriers uniform
  const int nBase = blockIdx.y * 32;
  const h16* arow = A + (size_t)(mBase + (lane & 15)) * K;
  const int kb = (lane < 16) ? 0 : 8;
  const int srow = tid >> 2, scol = (tid & 3) * 8;

  cp16B_to_lds(Bp + (size_t)srow * N + nBase + scol, &panel[0][srow][scol]);

  v8f acc0 = {}, acc1 = {};
  int s = 0;
  for (int k0 = 0; k0 < K; k0 += 32, s ^= 1) {
    wait_cp();
    __syncthreads();
    if (k0 + 32 < K)
      cp16B_to_lds(Bp + (size_t)(k0 + 32 + srow) * N + nBase + scol,
                   &panel[s ^ 1][srow][scol]);
    __builtin_prefetch(arow + k0 + 64, 0, 1);   // -> global_prefetch_b8
    v16h a = cat8(*(const v8h*)(arow + k0 + kb),
                  *(const v8h*)(arow + k0 + kb + 16));
    v16h b0, b1;
#pragma unroll
    for (int e = 0; e < 16; ++e) {
      int kr = kb + ((e < 8) ? e : (8 + e));
      b0[e] = panel[s][kr][lane & 15];
      b1[e] = panel[s][kr][16 + (lane & 15)];
    }
    acc0 = __builtin_amdgcn_wmma_f32_16x16x32_f16(false, a, false, b0,
                                                  (short)0, acc0, false, false);
    acc1 = __builtin_amdgcn_wmma_f32_16x16x32_f16(false, a, false, b1,
                                                  (short)0, acc1, false, false);
  }
  const int mrow = mBase + ((lane >> 4) << 3);
#pragma unroll
  for (int t = 0; t < 2; ++t) {
    const int n = nBase + 16 * t + (lane & 15);
    v8f acc = t ? acc1 : acc0;
#pragma unroll
    for (int r = 0; r < 8; ++r) {
      int   mm = mrow + r;
      float v  = acc[r];
      if (ep == EP_BIAS || ep == EP_BIAS_GELU) v += bias[n];
      if (ep == EP_BIAS_GELU) v = geluf(v);
      if (ep == EP_FILM_SIN) {
        int   bi = mm / rows_per_b;
        float ga = film[(size_t)bi * (2 * SHD) + n];
        float be = film[(size_t)bi * (2 * SHD) + SHD + n];
        v = sinf(OMEGA0 * ((v + bias[n]) * (1.0f + ga) + be));
      }
      if (C)  C[(size_t)mm * N + n]  = v;
      if (Ch) Ch[(size_t)mm * N + n] = (h16)v;
    }
  }
}

// ------------------------------- GroupNorm ---------------------------------
__global__ void gn_stats(const float* __restrict__ x, float* __restrict__ st,
                         int C, int HW, int g)
{
  const int bg = blockIdx.x, b = bg / g, gi = bg % g;
  const int Cg = C / g;
  const size_t base = ((size_t)b * C + (size_t)gi * Cg) * HW;
  const int n = Cg * HW;
  float s = 0.f, ss = 0.f;
  for (int i = threadIdx.x; i < n; i += blockDim.x) {
    float v = x[base + i]; s += v; ss += v * v;
  }
  __shared__ float sh[512];
  sh[threadIdx.x] = s; sh[256 + threadIdx.x] = ss;
  __syncthreads();
  for (int o = 128; o > 0; o >>= 1) {
    if ((int)threadIdx.x < o) {
      sh[threadIdx.x] += sh[threadIdx.x + o];
      sh[256 + threadIdx.x] += sh[256 + threadIdx.x + o];
    }
    __syncthreads();
  }
  if (threadIdx.x == 0) {
    float mu  = sh[0] / n;
    float var = sh[256] / n - mu * mu;
    st[2 * bg]     = mu;
    st[2 * bg + 1] = rsqrtf(var + 1e-5f);
  }
}

__global__ void gn_apply(const float* __restrict__ x, const float* __restrict__ st,
                         const float* __restrict__ gamma, const float* __restrict__ beta,
                         const float* __restrict__ resid,
                         float* __restrict__ o32, h16* __restrict__ o16,
                         int C, int HW, int g, int total, int doGelu)
{
  int i = blockIdx.x * blockDim.x + threadIdx.x;
  if (i >= total) return;
  int t = i / HW, c = t % C, b = t / C;
  int gi = c / (C / g);
  float mu = st[2 * (b * g + gi)], is = st[2 * (b * g + gi) + 1];
  float v = (x[i] - mu) * is * gamma[c] + beta[c];
  if (doGelu) v = geluf(v);
  if (resid)  v += resid[i];
  if (o32) o32[i] = v;
  if (o16) o16[i] = (h16)v;
}

// ------------------- encoder residual tail: gelu(a+b) -> concat f16 --------
__global__ void add_gelu_concat(const float* __restrict__ a, const float* __restrict__ b,
                                h16* __restrict__ out16, int C, int HW, int Ctot,
                                int coff, int total)
{
  int i = blockIdx.x * blockDim.x + threadIdx.x;
  if (i >= total) return;
  float v = geluf(a[i] + b[i]);
  int hw = i % HW, t = i / HW, c = t % C, bb = t / C;
  out16[((size_t)bb * Ctot + coff + c) * HW + hw] = (h16)v;
}

__global__ void k_add(float* __restrict__ a, const float* __restrict__ b, int n)
{
  int i = blockIdx.x * blockDim.x + threadIdx.x;
  if (i < n) a[i] += b[i];
}

__global__ void cvt_f16(const float* __restrict__ in, h16* __restrict__ out, int n)
{
  int i = blockIdx.x * blockDim.x + threadIdx.x;
  if (i < n) out[i] = (h16)in[i];
}

// ------------------------------ weight packing -----------------------------
__global__ void pack_conv_w(const float* __restrict__ w, h16* __restrict__ out,
                            int Cout, int K, int Kpad)
{ // w: [Cout][K] flat; out: [Kpad][Cout]
  int i = blockIdx.x * blockDim.x + threadIdx.x;
  if (i >= Kpad * Cout) return;
  int kk = i / Cout, co = i % Cout;
  out[i] = (h16)((kk < K) ? w[(size_t)co * K + kk] : 0.0f);
}

__global__ void pack_gemm_w(const float* __restrict__ w, h16* __restrict__ out,
                            int N, int K)
{ // w: [N][K] row-major (y = x @ w.T); out: [K][N]
  int i = blockIdx.x * blockDim.x + threadIdx.x;
  if (i >= N * K) return;
  int k = i / N, n = i % N;
  out[i] = (h16)w[(size_t)n * K + k];
}

// ------------------------ x_mag: 128x128 DFT magnitude ---------------------
__global__ void build_dft128(float* __restrict__ tw)
{
  int t = blockIdx.x * blockDim.x + threadIdx.x;
  if (t >= 128) return;
  float ang = -2.0f * PI32 * (float)t / 128.0f;
  tw[2 * t] = cosf(ang); tw[2 * t + 1] = sinf(ang);
}

__global__ void dft128_w(const float* __restrict__ x, const float* __restrict__ tw,
                         float* __restrict__ out)
{
  int i = blockIdx.x * blockDim.x + threadIdx.x;
  if (i >= BN * 128 * 128) return;
  int col = i % 128, y = (i / 128) % 128, b = i / (128 * 128);
  const float* row = x + ((size_t)b * 128 + y) * 128;
  float re = 0.f, im = 0.f;
  for (int w = 0; w < 128; ++w) {
    int t = (w * col) & 127;
    float v = row[w];
    re += v * tw[2 * t]; im += v * tw[2 * t + 1];
  }
  out[2 * (size_t)i] = re; out[2 * (size_t)i + 1] = im;
}

__global__ void dft128_h_mag(const float* __restrict__ X, const float* __restrict__ tw,
                             h16* __restrict__ mag16)
{
  int i = blockIdx.x * blockDim.x + threadIdx.x;
  if (i >= BN * 128 * 128) return;
  int col = i % 128, ky = (i / 128) % 128, b = i / (128 * 128);
  float re = 0.f, im = 0.f;
  for (int y = 0; y < 128; ++y) {
    int t = (ky * y) & 127;
    size_t xi = (((size_t)b * 128 + y) * 128 + col) * 2;
    float a = X[xi], bb = X[xi + 1];
    float c = tw[2 * t], s = tw[2 * t + 1];   // e^{-i2pi t/128} = c + i s
    re += a * c - bb * s; im += a * s + bb * c;
  }
  float m = sqrtf(re * re + im * im) * (1.0f / 128.0f);
  mag16[i] = (h16)log1pf(m);
}

// --------------------------- FNO 32-point transforms -----------------------
DEVFN void tw32_fill(float* c, float* s)
{
  for (int t = threadIdx.x; t < 32; t += blockDim.x) {
    float ang = -2.0f * PI32 * (float)t / 32.0f;
    c[t] = cosf(ang); s[t] = sinf(ang);   // e^{-i2pi t/32} = c + i s
  }
  __syncthreads();
}

__global__ void fno_fft_w(const float* __restrict__ x, float* __restrict__ out)
{
  __shared__ float c[32], s[32]; tw32_fill(c, s);
  int i = blockIdx.x * blockDim.x + threadIdx.x;
  if (i >= BN * WIDTH * 32 * 16) return;
  int col = i & 15, y = (i >> 4) & 31, bc = i >> 9;
  const float* row = x + ((size_t)bc * 32 + y) * 32;
  float re = 0.f, im = 0.f;
  for (int w = 0; w < 32; ++w) { int t = (w * col) & 31; re += row[w] * c[t]; im += row[w] * s[t]; }
  out[2 * (size_t)i] = re; out[2 * (size_t)i + 1] = im;
}

__global__ void fno_fft_h(const float* __restrict__ X, float* __restrict__ out)
{
  __shared__ float c[32], s[32]; tw32_fill(c, s);
  int i = blockIdx.x * blockDim.x + threadIdx.x;
  if (i >= BN * WIDTH * 32 * 16) return;
  int col = i & 15, ky = (i >> 4) & 31, bc = i >> 9;
  float re = 0.f, im = 0.f;
  for (int y = 0; y < 32; ++y) {
    int t = (ky * y) & 31;
    size_t xi = (((size_t)bc * 32 + y) * 16 + col) * 2;
    float a = X[xi], bb = X[xi + 1];
    re += a * c[t] - bb * s[t]; im += a * s[t] + bb * c[t];
  }
  out[2 * (size_t)i] = re * (1.0f / 32.0f);
  out[2 * (size_t)i + 1] = im * (1.0f / 32.0f);
}

__global__ void fno_spec1(const float* __restrict__ xf, const float* __restrict__ w1in,
                          const float* __restrict__ w2in, float* __restrict__ G)
{
  int i = blockIdx.x * blockDim.x + threadIdx.x;
  if (i >= BN * 16 * 32 * 16) return;
  int col = i & 15, ky = (i >> 4) & 31, r = (i >> 9) & 15, b = i >> 13;
  const float* W = (ky < 16) ? w1in : w2in;
  int x = (ky < 16) ? ky : ky - 16;
  float re = 0.f, im = 0.f;
  for (int ci = 0; ci < WIDTH; ++ci) {
    size_t xi = ((((size_t)b * WIDTH + ci) * 32 + ky) * 16 + col) * 2;
    size_t wi = ((((size_t)ci * 16 + r) * 16 + x) * 16 + col) * 2;
    float a = xf[xi], bb = xf[xi + 1], wc = W[wi], ws = W[wi + 1];
    re += a * wc - bb * ws; im += a * ws + bb * wc;
  }
  G[2 * (size_t)i] = re; G[2 * (size_t)i + 1] = im;
}

__global__ void fno_spec2(const float* __restrict__ G, const float* __restrict__ w1out,
                          const float* __restrict__ w2out, float* __restrict__ OF)
{
  int i = blockIdx.x * blockDim.x + threadIdx.x;
  if (i >= BN * WIDTH * 32 * 16) return;
  int col = i & 15, ky = (i >> 4) & 31, o = (i >> 9) & 63, b = i >> 15;
  const float* W = (ky < 16) ? w1out : w2out;
  int x = (ky < 16) ? ky : ky - 16;
  float re = 0.f, im = 0.f;
  for (int r = 0; r < 16; ++r) {
    size_t gi = ((((size_t)b * 16 + r) * 32 + ky) * 16 + col) * 2;
    size_t wi = ((((size_t)r * WIDTH + o) * 16 + x) * 16 + col) * 2;
    float a = G[gi], bb = G[gi + 1], wc = W[wi], ws = W[wi + 1];
    re += a * wc - bb * ws; im += a * ws + bb * wc;
  }
  OF[2 * (size_t)i] = re; OF[2 * (size_t)i + 1] = im;
}

__global__ void fno_idft_h(const float* __restrict__ OF, float* __restrict__ out)
{
  __shared__ float c[32], s[32]; tw32_fill(c, s);
  int i = blockIdx.x * blockDim.x + threadIdx.x;
  if (i >= BN * WIDTH * 32 * 16) return;
  int col = i & 15, y = (i >> 4) & 31, bc = i >> 9;
  float re = 0.f, im = 0.f;
  for (int ky = 0; ky < 32; ++ky) {
    int t = (ky * y) & 31;
    size_t xi = (((size_t)bc * 32 + ky) * 16 + col) * 2;
    float a = OF[xi], bb = OF[xi + 1];
    re += a * c[t] + bb * s[t];       // (a+ib)(c - i s)
    im += bb * c[t] - a * s[t];
  }
  out[2 * (size_t)i] = re; out[2 * (size_t)i + 1] = im;
}

__global__ void fno_irfft_w(const float* __restrict__ Y, float* __restrict__ out)
{
  __shared__ float c[32], s[32]; tw32_fill(c, s);
  int i = blockIdx.x * blockDim.x + threadIdx.x;
  if (i >= BN * WIDTH * 32 * 32) return;
  int w = i & 31, y = (i >> 5) & 31, bc = i >> 10;
  size_t base = (((size_t)bc * 32 + y) * 16) * 2;
  float v = Y[base];                            // col 0, real part
  for (int col = 1; col < 16; ++col) {
    int t = (col * w) & 31;
    float a = Y[base + 2 * col], bb = Y[base + 2 * col + 1];
    v += 2.0f * (a * c[t] + bb * s[t]);         // 2*Re((a+ib) e^{+i th})
  }
  out[(size_t)i] = v * (1.0f / 32.0f);
}

// ------------------------------- pooling / head ----------------------------
__global__ void pool_mean(const float* __restrict__ x, float* __restrict__ z,
                          h16* __restrict__ z16)
{
  int bc = blockIdx.x;
  const float* p = x + (size_t)bc * 1024;
  float s = 0.f;
  for (int i = threadIdx.x; i < 1024; i += 256) s += p[i];
  __shared__ float sh[256];
  sh[threadIdx.x] = s; __syncthreads();
  for (int o = 128; o > 0; o >>= 1) {
    if ((int)threadIdx.x < o) sh[threadIdx.x] += sh[threadIdx.x + o];
    __syncthreads();
  }
  if (threadIdx.x == 0) { float m = sh[0] / 1024.0f; z[bc] = m; z16[bc] = (h16)m; }
}

__global__ void siren_h0(const float* __restrict__ lam, const float* __restrict__ fw,
                         const float* __restrict__ fb, h16* __restrict__ h)
{
  int i = blockIdx.x * blockDim.x + threadIdx.x;
  if (i >= BN * OUT_LEN * SHD) return;
  int j = i % SHD, m = i / SHD, n = m % OUT_LEN;
  h[(size_t)i] = (h16)sinf(OMEGA0 * (lam[n] * fw[j] + fb[j]));
}

__global__ void out_head(const h16* __restrict__ h, const float* __restrict__ ow,
                         const float* __restrict__ ob, const float* __restrict__ sb,
                         float* __restrict__ out)
{
  int i = blockIdx.x * blockDim.x + threadIdx.x;
  if (i >= BN * 4 * OUT_LEN) return;
  int n = i % OUT_LEN, c = (i / OUT_LEN) % 4, b = i / (4 * OUT_LEN);
  const h16* hr = h + ((size_t)b * OUT_LEN + n) * SHD;
  const float* wr = ow + (size_t)c * SHD;
  float acc = 0.f;
  for (int k = 0; k < SHD; ++k) acc += (float)hr[k] * wr[k];
  float v = acc + ob[c] + sb[c * OUT_LEN + n];
  out[(size_t)i] = 1.0f / (1.0f + expf(-v));
}

// ===========================================================================
// Host orchestration
// ===========================================================================
extern "C" void kernel_launch(void* const* d_in, const int* in_sizes, int n_in,
                              void* d_out, int out_size, void* d_ws, size_t ws_size,
                              hipStream_t stream)
{
  (void)in_sizes; (void)out_size; (void)ws_size;
  // Leaf order: insertion-order flatten of setup_inputs() dict.
  //  0:x | 1..15 enc_real | 16..30 enc_k | 31..33 fuse {w,gw,gb}
  //  34..61 fno[4] x {w1_in,w1_out,w2_in,w2_out,bypass,gw,gb}
  //  62 to_z_w 63 to_z_b | 64 first_w 65 first_b | 66..71 hid_w | 72..77 hid_b
  //  78..83 film_w | 84..89 film_b | 90 out_w 91 out_b | 92 spec_bias 93 lam
  const float* L[96];
  for (int i = 0; i < n_in && i < 96; ++i) L[i] = (const float*)d_in[i];
  float* OUT = (float*)d_out;

  // ---- workspace arena (static offsets, deterministic) ----
  char* w8 = (char*)d_ws;
  size_t off = 0;
  auto arena = [&](size_t bytes) -> void* {
    void* p = w8 + off; off = (off + bytes + 255) & ~(size_t)255; return p;
  };
  h16* Pc[2][5];
  const int KpadC[5]  = {32, 288, 576, 576, 576};
  const int KrealC[5] = {9, 288, 576, 576, 576};
  const int CoutC[5]  = {32, 64, 64, 64, 64};
  for (int br = 0; br < 2; ++br)
    for (int j = 0; j < 5; ++j)
      Pc[br][j] = (h16*)arena((size_t)KpadC[j] * CoutC[j] * sizeof(h16));
  h16* Pfuse = (h16*)arena(128 * 64 * sizeof(h16));
  h16* Pbyp[NBLK];
  for (int i = 0; i < NBLK; ++i) Pbyp[i] = (h16*)arena(64 * 64 * sizeof(h16));
  h16* Ptoz = (h16*)arena(64 * 256 * sizeof(h16));
  h16* Pfilm[NLAY]; for (int l = 0; l < NLAY; ++l) Pfilm[l] = (h16*)arena(256 * 512 * sizeof(h16));
  h16* Phid[NLAY];  for (int l = 0; l < NLAY; ++l) Phid[l]  = (h16*)arena(256 * 256 * sizeof(h16));
  float* TW128   = (float*)arena(128 * 2 * sizeof(float));
  float* CB1     = (float*)arena((size_t)BN * 128 * 128 * 2 * sizeof(float));
  h16*   X16     = (h16*)arena((size_t)BN * 128 * 128 * sizeof(h16));
  h16*   XMAG16  = (h16*)arena((size_t)BN * 128 * 128 * sizeof(h16));
  float* CONVF   = (float*)arena((size_t)BN * 32 * 128 * 128 * sizeof(float));
  h16*   ACTA    = (h16*)arena((size_t)BN * 32 * 128 * 128 * sizeof(h16));
  h16*   ACTB    = (h16*)arena((size_t)BN * 64 * 64 * 64 * sizeof(h16));
  float* H3F32   = (float*)arena((size_t)BN * 64 * 1024 * sizeof(float));
  float* R2F32   = (float*)arena((size_t)BN * 64 * 1024 * sizeof(float));
  h16*   CONCAT  = (h16*)arena((size_t)BN * 128 * 1024 * sizeof(h16));
  float* FEAT    = (float*)arena((size_t)BN * 64 * 1024 * sizeof(float));
  h16*   FEAT16  = (h16*)arena((size_t)BN * 64 * 1024 * sizeof(h16));
  float* XFC     = (float*)arena((size_t)BN * 64 * 32 * 16 * 2 * sizeof(float));
  float* XHC     = (float*)arena((size_t)BN * 64 * 32 * 16 * 2 * sizeof(float));
  float* GBUF    = (float*)arena((size_t)BN * 16 * 32 * 16 * 2 * sizeof(float));
  float* SPEC    = (float*)arena((size_t)BN * 64 * 1024 * sizeof(float));
  float* BYP     = (float*)arena((size_t)BN * 64 * 1024 * sizeof(float));
  float* STATS   = (float*)arena((size_t)BN * 16 * 2 * sizeof(float));
  float* Z0      = (float*)arena((size_t)BN * 64 * sizeof(float));
  h16*   Z016    = (h16*)arena((size_t)BN * 64 * sizeof(h16));
  float* ZF32    = (float*)arena((size_t)BN * 256 * sizeof(float));
  h16*   Z16     = (h16*)arena((size_t)BN * 256 * sizeof(h16));
  float* FP      = (float*)arena((size_t)NLAY * BN * 512 * sizeof(float));
  h16*   HA      = (h16*)arena((size_t)BN * OUT_LEN * SHD * sizeof(h16));
  h16*   HB      = (h16*)arena((size_t)BN * OUT_LEN * SHD * sizeof(h16));

  auto g1 = [](int n) { return dim3((n + 255) / 256); };
  const dim3 B256(256);

  // ---- pack all weights ----
  for (int br = 0; br < 2; ++br) {
    int base = 1 + br * 15;
    const int widx[5] = {base + 0, base + 3, base + 6, base + 9, base + 12};
    for (int j = 0; j < 5; ++j)
      pack_conv_w<<<g1(KpadC[j] * CoutC[j]), B256, 0, stream>>>(
          L[widx[j]], Pc[br][j], CoutC[j], KrealC[j], KpadC[j]);
  }
  pack_conv_w<<<g1(128 * 64), B256, 0, stream>>>(L[31], Pfuse, 64, 128, 128);
  for (int i = 0; i < NBLK; ++i)
    pack_conv_w<<<g1(64 * 64), B256, 0, stream>>>(L[34 + 7 * i + 4], Pbyp[i], 64, 64, 64);
  pack_gemm_w<<<g1(64 * 256), B256, 0, stream>>>(L[62], Ptoz, 256, 64);
  for (int l = 0; l < NLAY; ++l) {
    pack_gemm_w<<<g1(256 * 512), B256, 0, stream>>>(L[78 + l], Pfilm[l], 512, 256);
    pack_gemm_w<<<g1(256 * 256), B256, 0, stream>>>(L[66 + l], Phid[l], 256, 256);
  }

  // ---- helpers ----
  auto conv = [&](const h16* in, const h16* wp, float* out,
                  int Cin, int H, int W, int Cout, int ks, int pad, int stride, int Kpad) {
    int Ho = (H + 2 * pad - ks) / stride + 1;
    int Wo = (W + 2 * pad - ks) / stride + 1;
    int tiles = (Ho * Wo) / 16;
    dim3 blk(32, 4), grd((tiles + 3) / 4, Cout / 32, BN);
    conv_wmma<<<grd, blk, 0, stream>>>(in, wp, out, Cin, H, W, Cout, Ho, Wo,
                                       ks, pad, stride, Kpad);
  };
  auto gn = [&](const float* x, const float* gw, const float* gb, const float* resid,
                float* o32, h16* o16, int C, int HW, int doGelu) {
    const int g = 16;
    gn_stats<<<dim3(BN * g), B256, 0, stream>>>(x, STATS, C, HW, g);
    int total = BN * C * HW;
    gn_apply<<<g1(total), B256, 0, stream>>>(x, STATS, gw, gb, resid, o32, o16,
                                             C, HW, g, total, doGelu);
  };
  auto gemm = [&](const h16* A, const h16* Bp, float* C, h16* Ch, int M, int N, int K,
                  const float* bias, const float* film, int rpb, int ep) {
    dim3 blk(32, 4), grd((M / 16 + 3) / 4, N / 32, 1);
    gemm_wmma<<<grd, blk, 0, stream>>>(A, Bp, C, Ch, M, N, K, bias, film, rpb, ep);
  };
  auto branch = [&](int b, const h16* in16, int coff) {
    conv(in16, Pc[coff ? 1 : 0][0], CONVF, 1, 128, 128, 32, 3, 1, 1, 32);
    gn(CONVF, L[b + 1], L[b + 2], nullptr, nullptr, ACTA, 32, 16384, 1);
    conv(ACTA, Pc[coff ? 1 : 0][1], CONVF, 32, 128, 128, 64, 3, 1, 2, 288);
    gn(CONVF, L[b + 4], L[b + 5], nullptr, nullptr, ACTB, 64, 4096, 1);
    conv(ACTB, Pc[coff ? 1 : 0][2], CONVF, 64, 64, 64, 64, 3, 1, 2, 576);
    gn(CONVF, L[b + 7], L[b + 8], nullptr, H3F32, ACTA, 64, 1024, 1);
    conv(ACTA, Pc[coff ? 1 : 0][3], CONVF, 64, 32, 32, 64, 3, 1, 1, 576);
    gn(CONVF, L[b + 10], L[b + 11], nullptr, nullptr, ACTB, 64, 1024, 1);
    conv(ACTB, Pc[coff ? 1 : 0][4], CONVF, 64, 32, 32, 64, 3, 1, 1, 576);
    gn(CONVF, L[b + 13], L[b + 14], nullptr, R2F32, nullptr, 64, 1024, 0);
    add_gelu_concat<<<g1(BN * 64 * 1024), B256, 0, stream>>>(
        R2F32, H3F32, CONCAT, 64, 1024, 128, coff, BN * 64 * 1024);
  };

  // ---- encoder ----
  cvt_f16<<<g1(BN * 16384), B256, 0, stream>>>(L[0], X16, BN * 16384);
  build_dft128<<<dim3(1), dim3(128), 0, stream>>>(TW128);
  dft128_w<<<g1(BN * 128 * 128), B256, 0, stream>>>(L[0], TW128, CB1);
  dft128_h_mag<<<g1(BN * 128 * 128), B256, 0, stream>>>(CB1, TW128, XMAG16);
  branch(1, X16, 0);
  branch(16, XMAG16, 64);

  // ---- fuse ----
  conv(CONCAT, Pfuse, CONVF, 128, 32, 32, 64, 1, 0, 1, 128);
  gn(CONVF, L[32], L[33], nullptr, FEAT, FEAT16, 64, 1024, 1);

  // ---- FNO blocks ----
  for (int i = 0; i < NBLK; ++i) {
    int fb = 34 + 7 * i;
    int nC = BN * WIDTH * 32 * 16;
    fno_fft_w<<<g1(nC), B256, 0, stream>>>(FEAT, XFC);
    fno_fft_h<<<g1(nC), B256, 0, stream>>>(XFC, XHC);
    fno_spec1<<<g1(BN * 16 * 32 * 16), B256, 0, stream>>>(XHC, L[fb + 0], L[fb + 2], GBUF);
    fno_spec2<<<g1(nC), B256, 0, stream>>>(GBUF, L[fb + 1], L[fb + 3], XFC);
    fno_idft_h<<<g1(nC), B256, 0, stream>>>(XFC, XHC);
    fno_irfft_w<<<g1(BN * WIDTH * 1024), B256, 0, stream>>>(XHC, SPEC);
    conv(FEAT16, Pbyp[i], BYP, 64, 32, 32, 64, 1, 0, 1, 64);
    k_add<<<g1(BN * 64 * 1024), B256, 0, stream>>>(SPEC, BYP, BN * 64 * 1024);
    gn(SPEC, L[fb + 5], L[fb + 6], FEAT, FEAT, FEAT16, 64, 1024, 1);
  }

  // ---- pool -> z -> FiLM params ----
  pool_mean<<<dim3(BN * 64), B256, 0, stream>>>(FEAT, Z0, Z016);
  gemm(Z016, Ptoz, ZF32, Z16, BN, 256, 64, L[63], nullptr, 0, EP_BIAS_GELU);
  for (int l = 0; l < NLAY; ++l)
    gemm(Z16, Pfilm[l], FP + (size_t)l * BN * 512, nullptr, BN, 512, 256,
         L[84 + l], nullptr, 0, EP_BIAS);

  // ---- SIREN decoder ----
  siren_h0<<<g1(BN * OUT_LEN * SHD), B256, 0, stream>>>(L[93], L[64], L[65], HA);
  h16* cur = HA; h16* nxt = HB;
  for (int l = 0; l < NLAY; ++l) {
    gemm(cur, Phid[l], nullptr, nxt, BN * OUT_LEN, SHD, SHD,
         L[72 + l], FP + (size_t)l * BN * 512, OUT_LEN, EP_FILM_SIN);
    h16* t = cur; cur = nxt; nxt = t;
  }

  // ---- head ----
  out_head<<<g1(BN * 4 * OUT_LEN), B256, 0, stream>>>(cur, L[90], L[91], L[92], OUT);
}
